// Model_71073118814925
// MI455X (gfx1250) — compile-verified, matrix operand order checked by reading, not run on hardware
//
#include <hip/hip_runtime.h>

// ---------------------------------------------------------------------------
// MI455X (gfx1250) optimal-stopping loss:
//   in_prod = ps @ l ; cdf = 0.5*(1 - tanh(40*cumsum(in_prod^2, T)))
//   out = -mean_n( sum_t pf * cdf )
//
// GEMM: split-bf16 V_WMMA_F32_16X16X32_BF16 (3 WMMAs/k-step ~ fp32 accuracy
// at the bf16 matrix-pipe rate; 8x FLOP/inst vs native f32 16x16x4 WMMA).
// B (l) is pre-packed ONCE into the exact per-lane WMMA B-operand tiling in
// workspace, so the hot loop feeds B with coalesced global_load_b128 from L2
// (4 MB packed array, L2-resident) -- no per-chunk LDS staging, no barriers
// inside the T loop. A (ps rows) is staged once per WG in LDS as hi/lo bf16
// with a padded stride for conflict-free ds_load_b128.
// Cumsum = 16-lane segmented scan matching the WMMA C layout; tanh folded
// into 1/(1+exp(80y)). Deterministic two-stage reduction (no float atomics).
// ---------------------------------------------------------------------------

typedef __bf16 bf16;
typedef __attribute__((ext_vector_type(16))) __bf16 v16bf;
typedef __attribute__((ext_vector_type(8)))  __bf16 bf16x8;
typedef __attribute__((ext_vector_type(4)))  __bf16 bf16x4;
typedef __attribute__((ext_vector_type(8)))  float  v8f;

#define N_ROWS   8192
#define DK       512
#define TCOLS    2048
#define ROWS_WG  32              // 2 waves x 16 rows each
#define THREADS  64
#define LDA      520             // padded LDS stride (bank-conflict-free b128)
#define NCHUNKS  (TCOLS / 16)    // 128
#define KSTEPS   (DK / 32)       // 16
#define NTUPLES  (NCHUNKS * KSTEPS * 2 * 32)   // 131072 b128 tiles per array

__device__ __forceinline__ v16bf cat8(bf16x8 a, bf16x8 b) {
  return __builtin_shufflevector(a, b, 0,1,2,3,4,5,6,7,8,9,10,11,12,13,14,15);
}

// ---------------------------------------------------------------------------
// One-time pack of l (512x2048 f32) into split-bf16 hi/lo arrays laid out in
// the per-lane WMMA B-operand order:
//   tuple t = (((chunk*16 + ks)*2 + g)*32 + lane), 8 contiguous K per tuple
//   lane: n = chunk*16 + (lane&15), K base = ks*32 + g*16 + 8*(lane>>4)
// Consecutive lanes -> consecutive 16B blocks => global_load_b128 coalesces.
// ---------------------------------------------------------------------------
__global__ __launch_bounds__(256) void pack_b_kernel(
    const float* __restrict__ l, bf16* __restrict__ Bh, bf16* __restrict__ Bl)
{
  const int t = blockIdx.x * 256 + threadIdx.x;      // < NTUPLES
  const int q = t & 31;
  const int g = (t >> 5) & 1;
  const int s = (t >> 6) & 15;
  const int j = t >> 10;
  const int n  = j * 16 + (q & 15);
  const int k0 = s * 32 + g * 16 + 8 * (q >> 4);
  bf16x8 hv, lv;
  #pragma unroll
  for (int e = 0; e < 8; ++e) {
    float x = l[(size_t)(k0 + e) * TCOLS + n];       // coalesced across lanes
    bf16 h = (bf16)x;
    hv[e] = h;
    lv[e] = (bf16)(x - (float)h);
  }
  ((bf16x8*)Bh)[t] = hv;                             // contiguous b128 store
  ((bf16x8*)Bl)[t] = lv;
}

// ---------------------------------------------------------------------------
// Main kernel (packed-B path): A from LDS, B straight from L2, zero barriers
// in the T loop.
// ---------------------------------------------------------------------------
__global__ __launch_bounds__(THREADS) void sigloss_packed(
    const float* __restrict__ ps, const float* __restrict__ pf,
    const bf16* __restrict__ Bh, const bf16* __restrict__ Bl,
    float* __restrict__ ws)
{
  __shared__ bf16 AsH[ROWS_WG * LDA];
  __shared__ bf16 AsL[ROWS_WG * LDA];

  const int tid     = threadIdx.x;
  const int lane    = tid & 31;
  const int wv      = tid >> 5;
  const int rowbase = blockIdx.x * ROWS_WG;

  // ---- stage A (ps rows) -> split bf16 hi/lo in LDS (once per WG) ----
  {
    const float4* ps4 = (const float4*)(ps + (size_t)rowbase * DK);
    for (int i = tid; i < ROWS_WG * DK / 4; i += THREADS) {
      float4 v = ps4[i];
      int e = i << 2;
      int r = e >> 9;
      int c = e & (DK - 1);
      int off = r * LDA + c;
      bf16 h0 = (bf16)v.x, h1 = (bf16)v.y, h2 = (bf16)v.z, h3 = (bf16)v.w;
      bf16x4 hv = {h0, h1, h2, h3};
      bf16x4 lv = {(bf16)(v.x - (float)h0), (bf16)(v.y - (float)h1),
                   (bf16)(v.z - (float)h2), (bf16)(v.w - (float)h3)};
      *(bf16x4*)&AsH[off] = hv;
      *(bf16x4*)&AsL[off] = lv;
    }
  }
  __syncthreads();   // the only barrier in the kernel

  const int m16   = lane & 15;
  const int hi16  = lane >> 4;
  const int aRow  = (wv * 16 + m16) * LDA;

  float carry[8];
  #pragma unroll
  for (int r = 0; r < 8; ++r) carry[r] = 0.f;
  float lossacc = 0.f;

  for (int chunk = 0; chunk < NCHUNKS; ++chunk) {
    v8f acc = {0.f,0.f,0.f,0.f,0.f,0.f,0.f,0.f};

    #pragma unroll 4
    for (int ks = 0; ks < KSTEPS; ++ks) {
      const int k0 = ks * 32 + hi16 * 8;
      v16bf ah = cat8(*(const bf16x8*)&AsH[aRow + k0],
                      *(const bf16x8*)&AsH[aRow + k0 + 16]);
      v16bf al = cat8(*(const bf16x8*)&AsL[aRow + k0],
                      *(const bf16x8*)&AsL[aRow + k0 + 16]);
      const size_t tb = ((size_t)(chunk * KSTEPS + ks) * 2 * 32) * 8;
      const bf16x8* bhp = (const bf16x8*)(Bh + tb) + lane;   // g=0 block
      const bf16x8* blp = (const bf16x8*)(Bl + tb) + lane;
      v16bf bh = cat8(bhp[0], bhp[32]);                      // +32 tiles = g=1
      v16bf bl = cat8(blp[0], blp[32]);
      acc = __builtin_amdgcn_wmma_f32_16x16x32_bf16(false, ah, false, bh, (short)0, acc, false, false);
      acc = __builtin_amdgcn_wmma_f32_16x16x32_bf16(false, ah, false, bl, (short)0, acc, false, false);
      acc = __builtin_amdgcn_wmma_f32_16x16x32_bf16(false, al, false, bh, (short)0, acc, false, false);
    }

    // ---- cumsum of squares along t, cdf, dot with pf ----
    const int col0 = chunk * 16;
    #pragma unroll
    for (int r = 0; r < 8; ++r) {
      float x = acc[r];
      float s = x * x;
      #pragma unroll
      for (int d = 1; d < 16; d <<= 1) {
        float t = __shfl_up(s, d, 16);
        if (m16 >= d) s += t;
      }
      float y = carry[r] + s;
      carry[r] = __shfl(y, 15, 16);
      float cdf = 1.0f / (1.0f + __expf(80.0f * y));   // 0.5*(1-tanh(40y))
      int row = rowbase + wv * 16 + r + 8 * hi16;
      lossacc += pf[(size_t)row * TCOLS + col0 + m16] * cdf;
    }
  }

  #pragma unroll
  for (int d = 16; d >= 1; d >>= 1) lossacc += __shfl_xor(lossacc, d, 32);
  if (lane == 0) ws[blockIdx.x * 2 + wv] = lossacc;
}

// ---------------------------------------------------------------------------
// Fallback (if workspace too small for packed B): per-chunk LDS staging.
// ---------------------------------------------------------------------------
#define LDB 520
__global__ __launch_bounds__(THREADS) void sigloss_staged(
    const float* __restrict__ l, const float* __restrict__ ps,
    const float* __restrict__ pf, float* __restrict__ ws)
{
  __shared__ bf16 AsH[ROWS_WG * LDA];
  __shared__ bf16 AsL[ROWS_WG * LDA];
  __shared__ bf16 BsH[16 * LDB];
  __shared__ bf16 BsL[16 * LDB];

  const int tid     = threadIdx.x;
  const int lane    = tid & 31;
  const int wv      = tid >> 5;
  const int rowbase = blockIdx.x * ROWS_WG;

  {
    const float4* ps4 = (const float4*)(ps + (size_t)rowbase * DK);
    for (int i = tid; i < ROWS_WG * DK / 4; i += THREADS) {
      float4 v = ps4[i];
      int e = i << 2;
      int r = e >> 9;
      int c = e & (DK - 1);
      int off = r * LDA + c;
      bf16 h0 = (bf16)v.x, h1 = (bf16)v.y, h2 = (bf16)v.z, h3 = (bf16)v.w;
      bf16x4 hv = {h0, h1, h2, h3};
      bf16x4 lv = {(bf16)(v.x - (float)h0), (bf16)(v.y - (float)h1),
                   (bf16)(v.z - (float)h2), (bf16)(v.w - (float)h3)};
      *(bf16x4*)&AsH[off] = hv;
      *(bf16x4*)&AsL[off] = lv;
    }
  }
  __syncthreads();

  const int m16  = lane & 15;
  const int hi16 = lane >> 4;

  float carry[8];
  #pragma unroll
  for (int r = 0; r < 8; ++r) carry[r] = 0.f;
  float lossacc = 0.f;

  for (int chunk = 0; chunk < NCHUNKS; ++chunk) {
    const int col0 = chunk * 16;
    {
      const int nq = tid & 3;
      for (int k = tid >> 2; k < DK; k += 16) {
        const float4 v = *(const float4*)&l[(size_t)k * TCOLS + col0 + nq * 4];
        const float xs[4] = {v.x, v.y, v.z, v.w};
        #pragma unroll
        for (int i = 0; i < 4; ++i) {
          const int n = nq * 4 + i;
          bf16 h = (bf16)xs[i];
          BsH[n * LDB + k] = h;
          BsL[n * LDB + k] = (bf16)(xs[i] - (float)h);
        }
      }
    }
    __syncthreads();

    v8f acc = {0.f,0.f,0.f,0.f,0.f,0.f,0.f,0.f};
    const int aRow = (wv * 16 + m16) * LDA;
    const int bRow = m16 * LDB;
    #pragma unroll 4
    for (int ks = 0; ks < KSTEPS; ++ks) {
      const int k0 = ks * 32 + hi16 * 8;
      v16bf ah = cat8(*(const bf16x8*)&AsH[aRow + k0],
                      *(const bf16x8*)&AsH[aRow + k0 + 16]);
      v16bf al = cat8(*(const bf16x8*)&AsL[aRow + k0],
                      *(const bf16x8*)&AsL[aRow + k0 + 16]);
      v16bf bh = cat8(*(const bf16x8*)&BsH[bRow + k0],
                      *(const bf16x8*)&BsH[bRow + k0 + 16]);
      v16bf bl = cat8(*(const bf16x8*)&BsL[bRow + k0],
                      *(const bf16x8*)&BsL[bRow + k0 + 16]);
      acc = __builtin_amdgcn_wmma_f32_16x16x32_bf16(false, ah, false, bh, (short)0, acc, false, false);
      acc = __builtin_amdgcn_wmma_f32_16x16x32_bf16(false, ah, false, bl, (short)0, acc, false, false);
      acc = __builtin_amdgcn_wmma_f32_16x16x32_bf16(false, al, false, bh, (short)0, acc, false, false);
    }

    #pragma unroll
    for (int r = 0; r < 8; ++r) {
      float x = acc[r];
      float s = x * x;
      #pragma unroll
      for (int d = 1; d < 16; d <<= 1) {
        float t = __shfl_up(s, d, 16);
        if (m16 >= d) s += t;
      }
      float y = carry[r] + s;
      carry[r] = __shfl(y, 15, 16);
      float cdf = 1.0f / (1.0f + __expf(80.0f * y));
      int row = rowbase + wv * 16 + r + 8 * hi16;
      lossacc += pf[(size_t)row * TCOLS + col0 + m16] * cdf;
    }
    __syncthreads();
  }

  #pragma unroll
  for (int d = 16; d >= 1; d >>= 1) lossacc += __shfl_xor(lossacc, d, 32);
  if (lane == 0) ws[blockIdx.x * 2 + wv] = lossacc;
}

__global__ __launch_bounds__(256) void reduce_kernel(
    const float* __restrict__ ws, float* __restrict__ out, int n)
{
  __shared__ float sm[256];
  int tid = threadIdx.x;
  float s = 0.f;
  for (int i = tid; i < n; i += 256) s += ws[i];
  sm[tid] = s;
  __syncthreads();
  for (int off = 128; off > 0; off >>= 1) {
    if (tid < off) sm[tid] += sm[tid + off];
    __syncthreads();
  }
  if (tid == 0) out[0] = -sm[0] * (1.0f / (float)N_ROWS);
}

extern "C" void kernel_launch(void* const* d_in, const int* in_sizes, int n_in,
                              void* d_out, int out_size, void* d_ws, size_t ws_size,
                              hipStream_t stream) {
  const float* l  = (const float*)d_in[0];   // (512, 2048)
  const float* ps = (const float*)d_in[1];   // (8192, 512)
  const float* pf = (const float*)d_in[2];   // (8192, 2048)
  float* out      = (float*)d_out;           // scalar
  float* partials = (float*)d_ws;            // first 2 KB of workspace

  const int nblocks = N_ROWS / ROWS_WG;      // 256 workgroups
  const size_t packElems = (size_t)DK * TCOLS;          // per hi/lo array
  const size_t need = 4096 + 2 * packElems * sizeof(bf16);  // ~4 MB

  if (ws_size >= need) {
    bf16* Bh = (bf16*)((char*)d_ws + 4096);
    bf16* Bl = Bh + packElems;
    pack_b_kernel<<<NTUPLES / 256, 256, 0, stream>>>(l, Bh, Bl);
    sigloss_packed<<<nblocks, THREADS, 0, stream>>>(ps, pf, Bh, Bl, partials);
  } else {
    sigloss_staged<<<nblocks, THREADS, 0, stream>>>(l, ps, pf, partials);
  }
  reduce_kernel<<<1, 256, 0, stream>>>(partials, out, nblocks * 2);
  (void)in_sizes; (void)n_in; (void)out_size; (void)ws_size;
}
